// WindowAttention_14800457302060
// MI455X (gfx1250) — compile-verified, matrix operand order checked by reading, not run on hardware
//
#include <hip/hip_runtime.h>

// ---------------------------------------------------------------------------
// MI455X (gfx1250) window-attention pipeline.
// All GEMM-shaped work on v_wmma_f32_16x16x32_f16 (wave32), fp32 accumulate,
// f16 data movement. Conv/GEMM kernel uses 2x2 register tiling per wave
// (32 pixels x 32 outch) to double WMMA compute per byte of fragment traffic.
// ---------------------------------------------------------------------------

typedef __attribute__((ext_vector_type(16))) _Float16 v16h;
typedef __attribute__((ext_vector_type(8)))  _Float16 h8;
typedef __attribute__((ext_vector_type(8)))  float    v8f;

#define BB    4
#define HH    256
#define WWID  256
#define CDIM  192
#define PIXT  (BB * HH * WWID)     // 262144 pixels (== 4096 windows * 64 tokens)
#define NWIN  4096
#define NHEAD 6
#define HDIM  32
#define QKC   128                  // per-branch conv output channels (q:0-63, k:64-127)
#define VQC   320                  // fused v(192) + qk1(128) output channels

// ---------------------------------------------------------------------------
// elementwise fp32 -> f16 convert
// ---------------------------------------------------------------------------
__global__ void cvt_f16_kernel(const float* __restrict__ x,
                               _Float16* __restrict__ o, int n) {
    int t = blockIdx.x * blockDim.x + threadIdx.x;
    if (t < n) o[t] = (_Float16)x[t];
}

// OIHW fp32 weights -> f16, layout wT[oc][k] with k = ((ky*KS)+kx)*CIN + c
__global__ void cvt_weight_kernel(const float* __restrict__ w,
                                  _Float16* __restrict__ o,
                                  int KS, int CIN, int COUT) {
    int t = blockIdx.x * blockDim.x + threadIdx.x;
    int K = KS * KS * CIN;
    if (t >= COUT * K) return;
    int oc   = t / K;
    int k    = t - oc * K;
    int cell = k / CIN;
    int c    = k - cell * CIN;
    int ky   = cell / KS;
    int kx   = cell - ky * KS;
    int src  = ((oc * CIN + c) * KS + ky) * KS + kx;
    o[t] = (_Float16)w[src];
}

__global__ void concat_bias_kernel(const float* __restrict__ a, int na,
                                   const float* __restrict__ b, int nb,
                                   float* __restrict__ o) {
    int t = blockIdx.x * blockDim.x + threadIdx.x;
    if (t < na)           o[t] = a[t];
    else if (t < na + nb) o[t] = b[t - na];
}

// ---------------------------------------------------------------------------
// Generic im2col conv as WMMA GEMM with 2x2 tile per wave.
//   in : f16 NHWC [PIXT][CIN]
//   wT : f16 [COUT][K], K = KS*KS*CIN (multiple of 8)
//   out: [PIXT][COUT], f16 or f32
// Wave tile: 32 pixels x (16*NT) outch, NT=2 when COUT%32==0 else 1.
// Block = 8 waves -> 256 pixels per block.x, blockIdx.y = outch supertile.
// ---------------------------------------------------------------------------
template <int KS, int CIN, int COUT, int PAD, bool LRELU, typename OutT>
__global__ void __launch_bounds__(256)
conv_wmma_kernel(const _Float16* __restrict__ in,
                 const _Float16* __restrict__ wT,
                 const float*    __restrict__ bias,
                 OutT*           __restrict__ out) {
    constexpr int K      = KS * KS * CIN;
    constexpr int KSTEPS = (K + 31) / 32;
    constexpr int NT     = (COUT % 32 == 0) ? 2 : 1;

    const int lane   = threadIdx.x & 31;
    const int wave   = threadIdx.x >> 5;
    const int psuper = blockIdx.x * 8 + wave;   // 32-pixel supertile
    const int osuper = blockIdx.y;              // (16*NT)-outch supertile

    const int mrow   = lane & 15;               // A-row / B-col lane index
    const int hiHalf = (lane >> 4) & 1;
    const int c0     = hiHalf * 8;              // per-lane K sub-offset

    // A-row pixels for the two m-tiles
    int pixA[2], bA[2], yA[2], xA[2];
#pragma unroll
    for (int mi = 0; mi < 2; ++mi) {
        pixA[mi] = psuper * 32 + mi * 16 + mrow;
        bA[mi]   = pixA[mi] >> 16;              // / (256*256)
        yA[mi]   = (pixA[mi] >> 8) & 255;
        xA[mi]   = pixA[mi] & 255;
    }

    // B-col out-channels for the NT n-tiles
    int occ[2];
    const _Float16* wrow[2];
#pragma unroll
    for (int ni = 0; ni < NT; ++ni) {
        occ[ni]  = osuper * (16 * NT) + ni * 16 + mrow;
        wrow[ni] = wT + (size_t)occ[ni] * K;
    }

    v8f acc[2][2] = {};

    for (int ks = 0; ks < KSTEPS; ++ks) {
        const int kbase = ks * 32;
        v16h afrag[2], bfrag[2];
#pragma unroll
        for (int q = 0; q < 2; ++q) {
            const int k0 = kbase + c0 + q * 16;
            // ---- A fragments (one per m-tile) ----
#pragma unroll
            for (int mi = 0; mi < 2; ++mi) {
                h8 av = {};
                if (k0 < K) {
                    if (KS == 1) {
                        av = *reinterpret_cast<const h8*>(in + (size_t)pixA[mi] * CIN + k0);
                    } else {
                        const int cell = k0 / CIN;
                        const int cch  = k0 - cell * CIN;
                        const int ky   = cell / KS;
                        const int kx   = cell - ky * KS;
                        const int iy   = yA[mi] + ky - PAD;
                        const int ix   = xA[mi] + kx - PAD;
                        if ((unsigned)iy < (unsigned)HH && (unsigned)ix < (unsigned)WWID) {
                            const size_t p = ((size_t)bA[mi] * HH + iy) * WWID + ix;
                            av = *reinterpret_cast<const h8*>(in + p * CIN + cch);
                        }
                    }
                }
#pragma unroll
                for (int e = 0; e < 8; ++e) afrag[mi][q * 8 + e] = av[e];
            }
            // ---- B fragments (one per n-tile) ----
#pragma unroll
            for (int ni = 0; ni < NT; ++ni) {
                h8 bv = {};
                if (k0 < K) bv = *reinterpret_cast<const h8*>(wrow[ni] + k0);
#pragma unroll
                for (int e = 0; e < 8; ++e) bfrag[ni][q * 8 + e] = bv[e];
            }
        }
#pragma unroll
        for (int mi = 0; mi < 2; ++mi)
#pragma unroll
            for (int ni = 0; ni < NT; ++ni)
                acc[mi][ni] = __builtin_amdgcn_wmma_f32_16x16x32_f16(
                    false, afrag[mi], false, bfrag[ni], (short)0, acc[mi][ni],
                    false, false);
    }

#pragma unroll
    for (int ni = 0; ni < NT; ++ni) {
        const float bval = bias[occ[ni]];
#pragma unroll
        for (int mi = 0; mi < 2; ++mi) {
#pragma unroll
            for (int r = 0; r < 8; ++r) {
                float y = acc[mi][ni][r] + bval;
                if (LRELU) y = (y >= 0.f) ? y : 0.2f * y;
                const int pix = psuper * 32 + mi * 16 + r + hiHalf * 8;
                out[(size_t)pix * COUT + occ[ni]] = (OutT)y;
            }
        }
    }
}

// ---------------------------------------------------------------------------
// Window attention: one block (4 waves) per (window, head).
//   S = Q K^T (WMMA) -> scale + rel-pos bias + softmax (fp32) -> O = P V (WMMA)
// Branch buffers passed as (pointer, row-stride) since qk1 lives inside the
// fused v|qk1 buffer (stride 320) while qk3/qk5 have stride 128.
// ---------------------------------------------------------------------------
__global__ void __launch_bounds__(128)
attn_kernel(const _Float16* __restrict__ qk1, int s1,
            const _Float16* __restrict__ qk3, int s3,
            const _Float16* __restrict__ qk5, int s5,
            const _Float16* __restrict__ vbuf, int sv,
            const float*    __restrict__ rpb,    // [225][6]
            _Float16*       __restrict__ ao) {   // [NWIN*64][192]
    __shared__ __align__(16) float    Sf[64 * 64];
    __shared__ __align__(16) _Float16 Ph[64 * 64];
    __shared__ __align__(16) _Float16 Vh[64 * 32];

    const int blk  = blockIdx.x;
    const int win  = blk / NHEAD;
    const int head = blk % NHEAD;

    const int b  = win >> 10;
    const int wi = win & 1023;
    const int wy = wi >> 5;
    const int wx = wi & 31;

    const int tid    = threadIdx.x;
    const int lane   = tid & 31;
    const int wave   = tid >> 5;
    const int mrow   = lane & 15;
    const int hiHalf = (lane >> 4) & 1;
    const int c0     = hiHalf * 8;

    auto tokpix = [&](int n) -> size_t {
        const int y = wy * 8 + (n >> 3);
        const int x = wx * 8 + (n & 7);
        return ((size_t)b * HH + y) * WWID + x;
    };

    const _Float16* qkb;
    int qs;
    if (head < 2)      { qkb = qk1; qs = s1; }
    else if (head < 4) { qkb = qk3; qs = s3; }
    else               { qkb = qk5; qs = s5; }
    const int qbase = (head & 1) * 32;        // q channels in branch buffer
    const int kbase = 64 + (head & 1) * 32;   // k channels in branch buffer

    // stage V head-slice into LDS: Vh[token][hd]
    for (int i = tid; i < 64 * 32; i += 128) {
        const int n = i >> 5, d = i & 31;
        Vh[i] = vbuf[tokpix(n) * sv + head * HDIM + d];
    }

    // ---- S = Q K^T : 16 tiles (4x4), 4 per wave, K = 32 (one WMMA each) ----
#pragma unroll
    for (int t = 0; t < 4; ++t) {
        const int tile = wave * 4 + t;
        const int mt = tile >> 2;
        const int nt = tile & 3;
        const size_t pq = tokpix(mt * 16 + mrow);
        const size_t pk = tokpix(nt * 16 + mrow);
        v16h a, bm;
#pragma unroll
        for (int q = 0; q < 2; ++q) {
            const int k0 = c0 + q * 16;
            h8 av = *reinterpret_cast<const h8*>(qkb + pq * qs + qbase + k0);
            h8 bv = *reinterpret_cast<const h8*>(qkb + pk * qs + kbase + k0);
#pragma unroll
            for (int e = 0; e < 8; ++e) { a[q * 8 + e] = av[e]; bm[q * 8 + e] = bv[e]; }
        }
        v8f acc = {};
        acc = __builtin_amdgcn_wmma_f32_16x16x32_f16(false, a, false, bm,
                                                     (short)0, acc, false, false);
#pragma unroll
        for (int r = 0; r < 8; ++r) {
            const int m = mt * 16 + r + hiHalf * 8;
            const int n = nt * 16 + mrow;
            Sf[m * 64 + n] = acc[r];
        }
    }
    __syncthreads();

    // ---- softmax over rows (scale + relative position bias) ----
    const float scale = 0.17677669529663687f;  // 32^-0.5
    if (tid < 64) {
        const int r  = tid;
        const int ry = r >> 3, rx = r & 7;
        float mx = -3.4e38f;
#pragma unroll 1
        for (int j = 0; j < 64; ++j) {
            const int dy = ry - (j >> 3) + 7;
            const int dx = rx - (j & 7) + 7;
            const float s = Sf[r * 64 + j] * scale + rpb[(dy * 15 + dx) * NHEAD + head];
            Sf[r * 64 + j] = s;
            mx = fmaxf(mx, s);
        }
        float sum = 0.f;
#pragma unroll 1
        for (int j = 0; j < 64; ++j) {
            const float e = __expf(Sf[r * 64 + j] - mx);
            Sf[r * 64 + j] = e;
            sum += e;
        }
        const float inv = 1.f / sum;
#pragma unroll 1
        for (int j = 0; j < 64; ++j)
            Ph[r * 64 + j] = (_Float16)(Sf[r * 64 + j] * inv);
    }
    __syncthreads();

    // ---- O = P V : 8 tiles (4 mtiles x 2 ntiles), 2 per wave, K = 64 ----
#pragma unroll
    for (int t = 0; t < 2; ++t) {
        const int tile = wave * 2 + t;
        const int mt = tile >> 1;
        const int nt = tile & 1;
        v8f acc = {};
#pragma unroll
        for (int ks = 0; ks < 2; ++ks) {
            const int kb = ks * 32;
            v16h a, bm;
#pragma unroll
            for (int q = 0; q < 2; ++q) {
                const int k0 = kb + c0 + q * 16;
                h8 av = *reinterpret_cast<const h8*>(&Ph[(mt * 16 + mrow) * 64 + k0]);
#pragma unroll
                for (int e = 0; e < 8; ++e) {
                    a[q * 8 + e]  = av[e];
                    bm[q * 8 + e] = Vh[(k0 + e) * 32 + nt * 16 + mrow];
                }
            }
            acc = __builtin_amdgcn_wmma_f32_16x16x32_f16(false, a, false, bm,
                                                         (short)0, acc, false, false);
        }
#pragma unroll
        for (int r = 0; r < 8; ++r) {
            const int m = mt * 16 + r + hiHalf * 8;    // token
            const int n = nt * 16 + mrow;              // hd
            ao[((size_t)win * 64 + m) * CDIM + head * HDIM + n] = (_Float16)acc[r];
        }
    }
}

// ---------------------------------------------------------------------------
// host-side orchestration
// ---------------------------------------------------------------------------
static inline size_t align256(size_t x) { return (x + 255) & ~(size_t)255; }

extern "C" void kernel_launch(void* const* d_in, const int* in_sizes, int n_in,
                              void* d_out, int out_size, void* d_ws, size_t ws_size,
                              hipStream_t stream) {
    (void)in_sizes; (void)n_in; (void)out_size; (void)ws_size;

    const float* x      = (const float*)d_in[0];
    const float* v_w    = (const float*)d_in[1];
    const float* v_b    = (const float*)d_in[2];
    const float* qk1_w  = (const float*)d_in[3];
    const float* qk1_b  = (const float*)d_in[4];
    const float* qk3_w1 = (const float*)d_in[5];
    const float* qk3_b1 = (const float*)d_in[6];
    const float* qk3_w2 = (const float*)d_in[7];
    const float* qk3_b2 = (const float*)d_in[8];
    const float* qk3_w3 = (const float*)d_in[9];
    const float* qk3_b3 = (const float*)d_in[10];
    const float* qk5_w1 = (const float*)d_in[11];
    const float* qk5_b1 = (const float*)d_in[12];
    const float* qk5_w2 = (const float*)d_in[13];
    const float* qk5_b2 = (const float*)d_in[14];
    const float* qk5_w3 = (const float*)d_in[15];
    const float* qk5_b3 = (const float*)d_in[16];
    const float* rpb    = (const float*)d_in[17];
    const float* proj_w = (const float*)d_in[18];
    const float* proj_b = (const float*)d_in[19];

    char*  ws  = (char*)d_ws;
    size_t off = 0;
    auto allocB = [&](size_t bytes) -> void* {
        void* p = ws + off;
        off = align256(off + bytes);
        return p;
    };
    auto allocH = [&](size_t elems) -> _Float16* {
        return (_Float16*)allocB(elems * sizeof(_Float16));
    };

    _Float16* xh   = allocH((size_t)PIXT * CDIM);   // f16 input (reused as ao later)
    _Float16* cbuf = allocH((size_t)PIXT * VQC);    // fused v(0..191) | qk1(192..319)
    _Float16* q3b  = allocH((size_t)PIXT * QKC);    // qk3 branch out
    _Float16* q5b  = allocH((size_t)PIXT * QKC);    // qk5 branch out
    _Float16* ta   = allocH((size_t)PIXT * 32);     // branch temp A
    _Float16* tb   = allocH((size_t)PIXT * 32);     // branch temp B
    _Float16* wvq  = allocH((size_t)VQC * 192);     // fused v|qk1 weights
    _Float16* w31  = allocH(32 * 9 * 192);
    _Float16* w32  = allocH(32 * 32);
    _Float16* w33  = allocH(128 * 9 * 32);
    _Float16* w51  = allocH(16 * 25 * 192);
    _Float16* w52  = allocH(16 * 16);
    _Float16* w53  = allocH(128 * 25 * 16);
    _Float16* wpj  = allocH(192 * 192);
    float*    bvq  = (float*)allocB(VQC * sizeof(float));
    _Float16* ao   = xh;                            // convs done before attention

    // ---- convert input + weights to f16 ----
    {
        const int n = PIXT * CDIM;
        cvt_f16_kernel<<<(n + 255) / 256, 256, 0, stream>>>(x, xh, n);
    }
    auto cw = [&](const float* src, _Float16* dst, int KS, int CIN, int COUT) {
        const int n = COUT * KS * KS * CIN;
        cvt_weight_kernel<<<(n + 255) / 256, 256, 0, stream>>>(src, dst, KS, CIN, COUT);
    };
    cw(v_w,    wvq,             1, 192, 192);   // rows   0..191 of fused weight
    cw(qk1_w,  wvq + 192 * 192, 1, 192, 128);   // rows 192..319 of fused weight
    cw(qk3_w1, w31, 3, 192, 32);
    cw(qk3_w2, w32, 1, 32,  32);
    cw(qk3_w3, w33, 3, 32,  128);
    cw(qk5_w1, w51, 5, 192, 16);
    cw(qk5_w2, w52, 1, 16,  16);
    cw(qk5_w3, w53, 5, 16,  128);
    cw(proj_w, wpj, 1, 192, 192);
    concat_bias_kernel<<<(VQC + 255) / 256, 256, 0, stream>>>(v_b, 192, qk1_b, 128, bvq);

    const dim3 cb(256);
    const int  gx = PIXT / 256;  // 8 waves/block * 32 pixels/wave

    // ---- fused v + qk1 (1x1, 192 -> 320) ----
    conv_wmma_kernel<1, 192, 320, 0, false, _Float16>
        <<<dim3(gx, 320 / 32), cb, 0, stream>>>(xh, wvq, bvq, cbuf);

    // ---- scale-3 branch: 3x3 -> lrelu -> 1x1 -> lrelu -> 3x3 ----
    conv_wmma_kernel<3, 192, 32, 1, true, _Float16>
        <<<dim3(gx, 32 / 32), cb, 0, stream>>>(xh, w31, qk3_b1, ta);
    conv_wmma_kernel<1, 32, 32, 0, true, _Float16>
        <<<dim3(gx, 32 / 32), cb, 0, stream>>>(ta, w32, qk3_b2, tb);
    conv_wmma_kernel<3, 32, 128, 1, false, _Float16>
        <<<dim3(gx, 128 / 32), cb, 0, stream>>>(tb, w33, qk3_b3, q3b);

    // ---- scale-5 branch: 5x5 -> lrelu -> 1x1 -> lrelu -> 5x5 ----
    conv_wmma_kernel<5, 192, 16, 2, true, _Float16>
        <<<dim3(gx, 16 / 16), cb, 0, stream>>>(xh, w51, qk5_b1, ta);
    conv_wmma_kernel<1, 16, 16, 0, true, _Float16>
        <<<dim3(gx, 16 / 16), cb, 0, stream>>>(ta, w52, qk5_b2, tb);
    conv_wmma_kernel<5, 16, 128, 2, false, _Float16>
        <<<dim3(gx, 128 / 32), cb, 0, stream>>>(tb, w53, qk5_b3, q5b);

    // ---- window attention (4096 windows x 6 heads) ----
    attn_kernel<<<NWIN * NHEAD, 128, 0, stream>>>(
        cbuf + 192, VQC,   // qk1 branch inside fused buffer
        q3b, QKC,
        q5b, QKC,
        cbuf, VQC,         // v inside fused buffer
        rpb, ao);

    // ---- output projection: (NWIN*64, 192) x (192, 192)^T + b -> fp32 out ----
    conv_wmma_kernel<1, 192, 192, 0, false, float>
        <<<dim3(gx, 192 / 32), cb, 0, stream>>>(ao, wpj, proj_b, (float*)d_out);
}
